// IpaScoreV5_60146722013704
// MI455X (gfx1250) — matrix operand order repaired and checked by reading, not compiled, not run on hardware
//
#include <hip/hip_runtime.h>
#include <hip/hip_bf16.h>

// ---------------------------------------------------------------------------
// Model constants (must match the reference)
// ---------------------------------------------------------------------------
enum {
  Bb_ = 2, Nn_ = 384, CS = 256, CZ = 128, CH = 256, Hh = 8,
  PQ = 8, PV = 12, NB = 4, FF = 256, THh = 4,
  BN = Bb_ * Nn_,                 // 768
  BNN = Bb_ * Nn_ * Nn_,          // 294912
  CAT = Hh * (CH + PV * 4 + CZ)   // 3456
};
static const float SCALE = 0.1f;

typedef __attribute__((ext_vector_type(16))) _Float16 v16h;
typedef __attribute__((ext_vector_type(8)))  _Float16 v8h;
typedef __attribute__((ext_vector_type(8)))  float    v8f;

// ---------------------------------------------------------------------------
// Device helpers
// ---------------------------------------------------------------------------
__device__ __forceinline__ float wave_sum(float v) {
  for (int o = 16; o > 0; o >>= 1) v += __shfl_xor(v, o, 32);
  return v;
}
__device__ __forceinline__ float wave_max(float v) {
  for (int o = 16; o > 0; o >>= 1) v = fmaxf(v, __shfl_xor(v, o, 32));
  return v;
}

// A fragment: row pointer -> two contiguous 8-half (16B) chunks -> v16h
// elems e<8:  K = hi*8 + e ; e>=8: K = 16 + hi*8 + (e-8)
__device__ __forceinline__ v16h frag_a(const _Float16* __restrict__ rowp, int hi) {
  v8h lo = *(const v8h*)(rowp + hi * 8);
  v8h hh = *(const v8h*)(rowp + 16 + hi * 8);
  return __builtin_shufflevector(lo, hh, 0,1,2,3,4,5,6,7,8,9,10,11,12,13,14,15);
}
// B fragment (from transposed tile): same K order as frag_a (colp + hi*8, colp + 16 + hi*8)
__device__ __forceinline__ v16h frag_b(const _Float16* __restrict__ colp, int hi) {
  v8h lo = *(const v8h*)(colp + hi * 8);
  v8h hh = *(const v8h*)(colp + 16 + hi * 8);
  return __builtin_shufflevector(lo, hh, 0,1,2,3,4,5,6,7,8,9,10,11,12,13,14,15);
}

// ---------------------------------------------------------------------------
// Generic WMMA GEMM:  C = epilogue( alpha * A @ B )
//   A: [M,K] row-major, strides (lda, batch outer sAo, inner sAi)
//   B: NN -> [K,N] row-major (ldb);  NT -> [N,K] row-major (ldb), computes A*B^T
//   epilogue: +bias[n], ReLU, +R (residual), * rmask[row]
// Tile: 32(M) x 64(N) x 32(K); 256 threads = 8 waves, one 16x16 WMMA tile/wave.
// LDS: A row-major (pad 40 halves), B transposed (pad 72 halves) so every
// fragment is 2x ds_load_b128, conflict-free.
// ---------------------------------------------------------------------------
enum { GF_BIAS = 1, GF_RELU = 2, GF_RES = 4, GF_NT = 8, GF_ROWMASK = 16 };

template <int FLAGS>
__global__ void __launch_bounds__(256)
k_gemm(const float* __restrict__ A, int lda, long sAo, long sAi,
       const float* __restrict__ Bm, int ldb, long sBo, long sBi,
       float* __restrict__ C, int ldc, long sCo, long sCi,
       const float* __restrict__ bias,
       const float* __restrict__ R, int ldr, long sRo, long sRi,
       const float* __restrict__ rmask,
       int M, int Ncols, int K, int innerB, float alpha)
{
  __shared__ __align__(16) _Float16 sA[32][40];    // [m][k]
  __shared__ __align__(16) _Float16 sBt[64][72];   // [n][k]  (transposed)

  const int bz = blockIdx.z;
  const int ob = bz / innerB, ib = bz % innerB;
  const float* Ab = A + ob * sAo + ib * sAi;
  const float* Bbp = Bm + ob * sBo + ib * sBi;
  float* Cb = C + ob * sCo + ib * sCi;
  const float* Rb = (FLAGS & GF_RES) ? (R + ob * sRo + ib * sRi) : nullptr;

  const int m0 = blockIdx.y * 32, n0 = blockIdx.x * 64;
  const int tid = threadIdx.x, lane = tid & 31, wv = tid >> 5;
  const int wm = (wv >> 2) * 16;   // 0 | 16
  const int wn = (wv & 3) * 16;    // 0,16,32,48
  const int hi = lane >> 4, l16 = lane & 15;

  v8f acc = {0.f, 0.f, 0.f, 0.f, 0.f, 0.f, 0.f, 0.f};

  for (int k0 = 0; k0 < K; k0 += 32) {
    // stage A tile (32m x 32k), row-major
    for (int idx = tid; idx < 32 * 32; idx += 256) {
      int r = idx >> 5, c = idx & 31;
      int gm = m0 + r, gk = k0 + c;
      float v = (gm < M && gk < K) ? Ab[(long)gm * lda + gk] : 0.f;
      sA[r][c] = (_Float16)v;
    }
    // stage B tile (32k x 64n) -> transposed sBt[n][k]
    for (int idx = tid; idx < 32 * 64; idx += 256) {
      int r = idx >> 6, c = idx & 63;    // r = k, c = n
      int gk = k0 + r, gn = n0 + c;
      float v = 0.f;
      if (gk < K && gn < Ncols)
        v = (FLAGS & GF_NT) ? Bbp[(long)gn * ldb + gk] : Bbp[(long)gk * ldb + gn];
      sBt[c][r] = (_Float16)v;
    }
    if (k0 + 32 < K) {  // gfx1250 speculative prefetch of next K-slab
      __builtin_prefetch(&Ab[(long)(m0 + (tid & 31)) * lda + k0 + 32], 0, 1);
      __builtin_prefetch((FLAGS & GF_NT) ? &Bbp[(long)(n0 + (tid & 63)) * ldb + k0 + 32]
                                         : &Bbp[(long)(k0 + 32 + (tid & 31)) * ldb + n0], 0, 1);
    }
    __syncthreads();

    v16h af = frag_a(&sA[wm + l16][0], hi);
    v16h bf = frag_b(&sBt[wn + l16][0], hi);
    acc = __builtin_amdgcn_wmma_f32_16x16x32_f16(false, af, false, bf,
                                                 (short)0, acc, false, false);
    __syncthreads();
  }

  const int gn = n0 + wn + l16;
  #pragma unroll
  for (int r = 0; r < 8; ++r) {
    int gm = m0 + wm + r + 8 * hi;
    if (gm < M && gn < Ncols) {
      float o = acc[r] * alpha;
      if (FLAGS & GF_BIAS) o += bias[gn];
      if (FLAGS & GF_RELU) o = fmaxf(o, 0.f);
      if (FLAGS & GF_RES)  o += Rb[(long)gm * ldr + gn];
      if (FLAGS & GF_ROWMASK) o *= rmask[gm];
      Cb[(long)gm * ldc + gn] = o;
    }
  }
}

// ---------------------------------------------------------------------------
// Fused edge transition (the dominant ~58 GFLOP of the net):
//   zz = z + left_i + right_j
//   z' = (zz + W2( relu( W1( LN(zz) ) + b1 ) ) + b2) * mask_i*mask_j
// One 32-row tile of z per block; W1/W2 staged (sequentially, transposed) in
// LDS f16. z double-buffered (zin -> zout); z stays L2-resident (151MB<192MB).
// ---------------------------------------------------------------------------
__global__ void __launch_bounds__(256)
k_edge_fused(const float* __restrict__ zin, float* __restrict__ zout,
             const float* __restrict__ leftb, const float* __restrict__ rightb,
             const float* __restrict__ lng, const float* __restrict__ lnb,
             const float* __restrict__ W1, const float* __restrict__ b1,
             const float* __restrict__ W2, const float* __restrict__ b2,
             const float* __restrict__ mask)
{
  __shared__ __align__(16) _Float16 sWt[128][136];  // [n][k] transposed weights
  __shared__ __align__(16) _Float16 sX[32][136];    // [m][k]
  __shared__ __align__(16) _Float16 sM[32][136];    // [m][k]

  const int tid = threadIdx.x, lane = tid & 31, wv = tid >> 5;
  const int hi = lane >> 4, l16 = lane & 15;
  const long row0 = (long)blockIdx.x * 32;
  const int nc = wv * 16 + l16;  // this wave's output column

  // ---- stage 1: zz = z + left + right ; LN -> sX (f16) ----
  for (int rr = wv; rr < 32; rr += 8) {
    long row = row0 + rr;
    int b = (int)(row / ((long)Nn_ * Nn_));
    long rem = row % ((long)Nn_ * Nn_);
    int i = (int)(rem / Nn_), j = (int)(rem % Nn_);
    const float* zr = zin + row * CZ;
    const float* lf = leftb + ((long)b * Nn_ + i) * CZ;
    const float* rg = rightb + ((long)b * Nn_ + j) * CZ;
    float v4[4], s = 0.f, ss = 0.f;
    #pragma unroll
    for (int t = 0; t < 4; ++t) {
      int c = lane + t * 32;
      float v = zr[c] + lf[c] + rg[c];
      v4[t] = v; s += v; ss += v * v;
    }
    s = wave_sum(s); ss = wave_sum(ss);
    float mu = s * (1.f / CZ), var = ss * (1.f / CZ) - mu * mu;
    float inv = rsqrtf(var + 1e-5f);
    #pragma unroll
    for (int t = 0; t < 4; ++t) {
      int c = lane + t * 32;
      sX[rr][c] = (_Float16)((v4[t] - mu) * inv * lng[c] + lnb[c]);
    }
  }
  // ---- load W1 transposed: sWt[n][k] = W1[k*128+n] ----
  for (int idx = tid; idx < 128 * 128; idx += 256)
    sWt[idx & 127][idx >> 7] = (_Float16)W1[idx];
  __syncthreads();

  // ---- GEMM1: (32x128) = sX @ W1, relu(+b1) -> sM ----
  v8f a0 = {0.f,0.f,0.f,0.f,0.f,0.f,0.f,0.f};
  v8f a1 = {0.f,0.f,0.f,0.f,0.f,0.f,0.f,0.f};
  for (int k0 = 0; k0 < 128; k0 += 32) {
    v16h bf = frag_b(&sWt[nc][k0], hi);
    v16h f0 = frag_a(&sX[l16][k0], hi);
    v16h f1 = frag_a(&sX[16 + l16][k0], hi);
    a0 = __builtin_amdgcn_wmma_f32_16x16x32_f16(false, f0, false, bf, (short)0, a0, false, false);
    a1 = __builtin_amdgcn_wmma_f32_16x16x32_f16(false, f1, false, bf, (short)0, a1, false, false);
  }
  {
    float bv = b1[nc];
    #pragma unroll
    for (int r = 0; r < 8; ++r) {
      int mr = r + 8 * hi;
      sM[mr][nc]      = (_Float16)fmaxf(a0[r] + bv, 0.f);
      sM[16 + mr][nc] = (_Float16)fmaxf(a1[r] + bv, 0.f);
    }
  }
  __syncthreads();
  // ---- reload W2 (transposed) over W1 ----
  for (int idx = tid; idx < 128 * 128; idx += 256)
    sWt[idx & 127][idx >> 7] = (_Float16)W2[idx];
  __syncthreads();

  // ---- GEMM2: (32x128) = sM @ W2 ----
  a0 = (v8f){0.f,0.f,0.f,0.f,0.f,0.f,0.f,0.f};
  a1 = (v8f){0.f,0.f,0.f,0.f,0.f,0.f,0.f,0.f};
  for (int k0 = 0; k0 < 128; k0 += 32) {
    v16h bf = frag_b(&sWt[nc][k0], hi);
    v16h f0 = frag_a(&sM[l16][k0], hi);
    v16h f1 = frag_a(&sM[16 + l16][k0], hi);
    a0 = __builtin_amdgcn_wmma_f32_16x16x32_f16(false, f0, false, bf, (short)0, a0, false, false);
    a1 = __builtin_amdgcn_wmma_f32_16x16x32_f16(false, f1, false, bf, (short)0, a1, false, false);
  }
  // ---- epilogue: recompute zz (L2-hot), add, mask, store ----
  float bv = b2[nc];
  #pragma unroll
  for (int g = 0; g < 2; ++g) {
    v8f& ac = g ? a1 : a0;
    #pragma unroll
    for (int r = 0; r < 8; ++r) {
      long row = row0 + g * 16 + r + 8 * hi;
      int b = (int)(row / ((long)Nn_ * Nn_));
      long rem = row % ((long)Nn_ * Nn_);
      int i = (int)(rem / Nn_), j = (int)(rem % Nn_);
      float zz = zin[row * CZ + nc]
               + leftb[((long)b * Nn_ + i) * CZ + nc]
               + rightb[((long)b * Nn_ + j) * CZ + nc];
      float em = mask[b * Nn_ + i] * mask[b * Nn_ + j];
      zout[row * CZ + nc] = (zz + ac[r] + bv) * em;
    }
  }
}

// ---------------------------------------------------------------------------
// LayerNorm: one wave32 per row
// ---------------------------------------------------------------------------
__global__ void __launch_bounds__(256)
k_layernorm(const float* __restrict__ x, const float* __restrict__ g,
            const float* __restrict__ b, float* __restrict__ y, int D, int rows)
{
  int wv = threadIdx.x >> 5, lane = threadIdx.x & 31;
  int row = blockIdx.x * 8 + wv;
  if (row >= rows) return;
  const float* xr = x + (long)row * D;
  float s = 0.f, ss = 0.f;
  for (int i = lane; i < D; i += 32) { float v = xr[i]; s += v; ss += v * v; }
  s = wave_sum(s); ss = wave_sum(ss);
  float mu = s / D, var = ss / D - mu * mu;
  float inv = rsqrtf(var + 1e-5f);
  float* yr = y + (long)row * D;
  for (int i = lane; i < D; i += 32) yr[i] = (xr[i] - mu) * inv * g[i] + b[i];
}

// ---------------------------------------------------------------------------
// Rigid frame helpers
// ---------------------------------------------------------------------------
__global__ void k_init_rigids(const float* __restrict__ rt, float* __restrict__ quat,
                              float* __restrict__ tr, int n) {
  int i = blockIdx.x * blockDim.x + threadIdx.x;
  if (i >= n) return;
  for (int c = 0; c < 4; ++c) quat[i * 4 + c] = rt[i * 7 + c];
  for (int c = 0; c < 3; ++c) tr[i * 3 + c] = rt[i * 7 + 4 + c] * SCALE;
}

__global__ void k_quat_rot(const float* __restrict__ quat, float* __restrict__ rot, int n) {
  int i = blockIdx.x * blockDim.x + threadIdx.x;
  if (i >= n) return;
  float w = quat[i*4], x = quat[i*4+1], y = quat[i*4+2], z = quat[i*4+3];
  float* r = rot + i * 9;
  r[0] = 1 - 2*(y*y + z*z); r[1] = 2*(x*y - w*z); r[2] = 2*(x*z + w*y);
  r[3] = 2*(x*y + w*z); r[4] = 1 - 2*(x*x + z*z); r[5] = 2*(y*z - w*x);
  r[6] = 2*(x*z - w*y); r[7] = 2*(y*z + w*x); r[8] = 1 - 2*(x*x + y*y);
}

__global__ void k_apply_rot(float* __restrict__ pts, const float* __restrict__ rot,
                            const float* __restrict__ tr, int pts_per_res, int total) {
  int i = blockIdx.x * blockDim.x + threadIdx.x;
  if (i >= total) return;
  int res = i / pts_per_res;
  float* p = pts + (long)i * 3;
  const float* R = rot + res * 9; const float* t = tr + res * 3;
  float a = p[0], b = p[1], c = p[2];
  p[0] = R[0]*a + R[1]*b + R[2]*c + t[0];
  p[1] = R[3]*a + R[4]*b + R[5]*c + t[1];
  p[2] = R[6]*a + R[7]*b + R[8]*c + t[2];
}

__global__ void k_pointnorm(const float* __restrict__ pts, float* __restrict__ out,
                            int P, int total) {
  int i = blockIdx.x * blockDim.x + threadIdx.x;
  if (i >= total) return;
  const float* p = pts + (long)i * P * 3;
  float s = 0.f;
  for (int j = 0; j < P * 3; ++j) s += p[j] * p[j];
  out[i] = s;
}

// loc = R^T (opt - tr) -> cat[...,2048:2336]; onorm -> cat[...,2336:2432]
__global__ void k_loc(const float* __restrict__ opt, const float* __restrict__ rot,
                      const float* __restrict__ tr, float* __restrict__ cat, int total) {
  int idx = blockIdx.x * blockDim.x + threadIdx.x;
  if (idx >= total) return;
  int res = idx / (Hh * PV), hp = idx % (Hh * PV);
  const float* p = opt + (long)idx * 3;
  const float* R = rot + res * 9; const float* t = tr + res * 3;
  float a = p[0]-t[0], b = p[1]-t[1], c = p[2]-t[2];
  float l0 = R[0]*a + R[3]*b + R[6]*c;
  float l1 = R[1]*a + R[4]*b + R[7]*c;
  float l2 = R[2]*a + R[5]*b + R[8]*c;
  float* cr = cat + (long)res * CAT;
  cr[Hh*CH + hp*3 + 0] = l0;
  cr[Hh*CH + hp*3 + 1] = l1;
  cr[Hh*CH + hp*3 + 2] = l2;
  cr[Hh*CH + Hh*PV*3 + hp] = sqrtf(l0*l0 + l1*l1 + l2*l2 + 1e-8f);
}

// ---------------------------------------------------------------------------
// IPA softmax: fuses qk, pair-bias, point-distance and mask terms.
// One wave per (b,h,i) row; N=384 = 12 regs/lane, fully in-register.
// ---------------------------------------------------------------------------
__global__ void __launch_bounds__(128)
k_ipa_softmax(float* __restrict__ attn, const float* __restrict__ pqk,
              const float* __restrict__ bwb, const float* __restrict__ qn,
              const float* __restrict__ kn, const float* __restrict__ gamma,
              const float* __restrict__ mask)
{
  int wv = threadIdx.x >> 5, lane = threadIdx.x & 31;
  int row = blockIdx.x * 4 + wv;                    // [0, B*H*N)
  int i = row % Nn_, bh = row / Nn_, h = bh % Hh, b = bh / Hh;
  float gm = gamma[h];
  float sp = (gm > 20.f) ? gm : log1pf(expf(gm));
  const float pw = 0.09622504486493763f;            // sqrt(1/(3*PQ*9/2))
  float coef = -0.5f * pw * sp;
  const float bbs = 0.5773502691896258f;            // sqrt(1/3)
  float qni = qn[((long)b * Nn_ + i) * Hh + h];
  long arow = (long)row * Nn_;
  float regs[12], mx = -1e30f;
  #pragma unroll
  for (int t = 0; t < 12; ++t) {
    int j = lane + t * 32;
    float a = attn[arow + j];
    float d2 = qni + kn[((long)b * Nn_ + j) * Hh + h] - 2.f * pqk[arow + j];
    float bbv = bwb[(((long)b * Nn_ + i) * Nn_ + j) * Hh + h] * bbs;
    float mb = (mask[b * Nn_ + j] > 0.f) ? 0.f : -1e9f;
    float lg = a + bbv + coef * d2 + mb;
    regs[t] = lg; mx = fmaxf(mx, lg);
  }
  mx = wave_max(mx);
  float sum = 0.f;
  #pragma unroll
  for (int t = 0; t < 12; ++t) { regs[t] = expf(regs[t] - mx); sum += regs[t]; }
  sum = wave_sum(sum);
  float inv = 1.f / sum;
  #pragma unroll
  for (int t = 0; t < 12; ++t) attn[arow + lane + t * 32] = regs[t] * inv;
}

// Transformer softmax (mask bias only); rows = B * heads * N
__global__ void __launch_bounds__(128)
k_softmax_mask(float* __restrict__ attn, const float* __restrict__ mask, int heads)
{
  int wv = threadIdx.x >> 5, lane = threadIdx.x & 31;
  int row = blockIdx.x * 4 + wv;
  int b = row / (heads * Nn_);
  long arow = (long)row * Nn_;
  float regs[12], mx = -1e30f;
  #pragma unroll
  for (int t = 0; t < 12; ++t) {
    int j = lane + t * 32;
    float mb = (mask[b * Nn_ + j] > 0.f) ? 0.f : -1e9f;
    float lg = attn[arow + j] + mb;
    regs[t] = lg; mx = fmaxf(mx, lg);
  }
  mx = wave_max(mx);
  float sum = 0.f;
  #pragma unroll
  for (int t = 0; t < 12; ++t) { regs[t] = expf(regs[t] - mx); sum += regs[t]; }
  sum = wave_sum(sum);
  float inv = 1.f / sum;
  #pragma unroll
  for (int t = 0; t < 12; ++t) attn[arow + lane + t * 32] = regs[t] * inv;
}

// ---------------------------------------------------------------------------
// Backbone update (one wave per residue): upd = ((s*diff)@W + b)*diff,
// quaternion normalize+compose, tr += R @ tv
// ---------------------------------------------------------------------------
__global__ void __launch_bounds__(256)
k_bb_update(const float* __restrict__ s, const float* __restrict__ mask,
            const float* __restrict__ fixedm, const float* __restrict__ W,
            const float* __restrict__ bias, const float* __restrict__ rot,
            float* __restrict__ quat, float* __restrict__ tr, int rows)
{
  int wv = threadIdx.x >> 5, lane = threadIdx.x & 31;
  int row = blockIdx.x * 8 + wv;
  if (row >= rows) return;
  float diff = (1.f - fixedm[row]) * mask[row];
  float c[6] = {0,0,0,0,0,0};
  for (int k = lane; k < CS; k += 32) {
    float sv = s[(long)row * CS + k] * diff;
    #pragma unroll
    for (int j = 0; j < 6; ++j) c[j] += sv * W[k * 6 + j];
  }
  #pragma unroll
  for (int j = 0; j < 6; ++j) c[j] = wave_sum(c[j]);
  if (lane == 0) {
    float upd[6];
    for (int j = 0; j < 6; ++j) upd[j] = (c[j] + bias[j]) * diff;
    float qw = 1.f, qx = upd[0], qy = upd[1], qz = upd[2];
    float nrm = sqrtf(qw*qw + qx*qx + qy*qy + qz*qz);
    qw /= nrm; qx /= nrm; qy /= nrm; qz /= nrm;
    const float* R = rot + row * 9;
    float tx = upd[3], ty = upd[4], tz = upd[5];
    tr[row*3+0] += R[0]*tx + R[1]*ty + R[2]*tz;
    tr[row*3+1] += R[3]*tx + R[4]*ty + R[5]*tz;
    tr[row*3+2] += R[6]*tx + R[7]*ty + R[8]*tz;
    float aw = quat[row*4], ax = quat[row*4+1], ay = quat[row*4+2], az = quat[row*4+3];
    quat[row*4+0] = aw*qw - ax*qx - ay*qy - az*qz;
    quat[row*4+1] = aw*qx + ax*qw + ay*qz - az*qy;
    quat[row*4+2] = aw*qy - ax*qz + ay*qw + az*qx;
    quat[row*4+3] = aw*qz + ax*qy - ay*qx + az*qw;
  }
}

// psi normalize + final7 pack
__global__ void k_final(const float* __restrict__ un, const float* __restrict__ quat,
                        const float* __restrict__ tr, float* __restrict__ out, int rows)
{
  int i = blockIdx.x * blockDim.x + threadIdx.x;
  if (i >= rows) return;
  float u0 = un[i*2], u1 = un[i*2+1];
  float n = sqrtf(fmaxf(u0*u0 + u1*u1, 1e-8f));
  out[i*2] = u0 / n; out[i*2+1] = u1 / n;
  float* f7 = out + (long)rows * 2 + (long)i * 7;
  for (int c = 0; c < 4; ++c) f7[c] = quat[i*4+c];
  for (int c = 0; c < 3; ++c) f7[4+c] = tr[i*3+c] / SCALE;
}

// ---------------------------------------------------------------------------
// Host-side GEMM dispatch
// ---------------------------------------------------------------------------
static void gemm(hipStream_t st, int flags,
                 const float* A, int lda, long sAo, long sAi,
                 const float* Bm, int ldb, long sBo, long sBi,
                 float* C, int ldc, long sCo, long sCi,
                 const float* bias,
                 const float* R, int ldr, long sRo, long sRi,
                 const float* rmask,
                 int M, int N, int K, int outer, int inner, float alpha)
{
  dim3 grid((N + 63) / 64, (M + 31) / 32, outer * inner);
#define GEMM_CASE(F)                                                              \
  case F:                                                                         \
    k_gemm<F><<<grid, 256, 0, st>>>(A, lda, sAo, sAi, Bm, ldb, sBo, sBi, C, ldc,  \
                                    sCo, sCi, bias, R, ldr, sRo, sRi, rmask, M,   \
                                    N, K, inner, alpha);                          \
    break;
  switch (flags) {
    GEMM_CASE(0) GEMM_CASE(1) GEMM_CASE(3) GEMM_CASE(5)
    GEMM_CASE(8) GEMM_CASE(20) GEMM_CASE(21)
    default: break;
  }
#undef GEMM_CASE
}

// Parameter pointers for one block (JAX sorted-dict flatten order)
struct BlockP {
  const float *bb_b, *bb_w;
  bool has_edge;
  const float *e_left_b, *e_left_w, *e_lns_b, *e_lns_g, *e_lnz_b, *e_lnz_g;
  const float *e_right_b, *e_right_w, *e_w1_b, *e_w1_w, *e_w2_b, *e_w2_w;
  const float *gamma, *ln_b, *ln_g, *wb, *wk, *wkp, *wo, *wq, *wqp, *wv, *wvp;
  const float *post_w;
  struct { const float *ln1_b,*ln1_g,*ln2_b,*ln2_g,*w1_b,*w1_w,*w2_b,*w2_w,*wo_b,*wo_w,*wqkv_b,*wqkv_w; } tf[2];
  const float *t_ln_b, *t_ln_g, *t_w1_b, *t_w1_w, *t_w2_b, *t_w2_w;
};

// ---------------------------------------------------------------------------
extern "C" void kernel_launch(void* const* d_in, const int* in_sizes, int n_in,
                              void* d_out, int out_size, void* d_ws, size_t ws_size,
                              hipStream_t stream)
{
  (void)in_sizes; (void)n_in; (void)out_size; (void)ws_size;
  // ---- inputs ----
  const float* edge_embed   = (const float*)d_in[1];  // (B,N,N,CZ)
  const float* res_mask     = (const float*)d_in[2];  // (B,N)
  const float* fixed_mask   = (const float*)d_in[3];  // (B,N)
  const float* rigids_embed = (const float*)d_in[4];  // (B,N,1,CS)
  const float* rigids_t     = (const float*)d_in[5];  // (B,N,1,7)

  // ---- param leaves in pytree flatten order ----
  int pi = 6;
  auto nxt = [&]() -> const float* { return (const float*)d_in[pi++]; };
  BlockP blk[NB];
  for (int b = 0; b < NB; ++b) {
    BlockP& P = blk[b];
    P.bb_b = nxt(); P.bb_w = nxt();
    P.has_edge = (b < NB - 1);
    if (P.has_edge) {
      P.e_left_b = nxt(); P.e_left_w = nxt();
      P.e_lns_b = nxt();  P.e_lns_g = nxt();
      P.e_lnz_b = nxt();  P.e_lnz_g = nxt();
      P.e_right_b = nxt(); P.e_right_w = nxt();
      P.e_w1_b = nxt(); P.e_w1_w = nxt();
      P.e_w2_b = nxt(); P.e_w2_w = nxt();
    }
    P.gamma = nxt(); P.ln_b = nxt(); P.ln_g = nxt();
    P.wb = nxt(); P.wk = nxt(); P.wkp = nxt(); P.wo = nxt();
    P.wq = nxt(); P.wqp = nxt(); P.wv = nxt(); P.wvp = nxt();
    P.post_w = nxt();
    for (int l = 0; l < 2; ++l) {
      auto& T = P.tf[l];
      T.ln1_b = nxt(); T.ln1_g = nxt(); T.ln2_b = nxt(); T.ln2_g = nxt();
      T.w1_b = nxt(); T.w1_w = nxt(); T.w2_b = nxt(); T.w2_w = nxt();
      T.wo_b = nxt(); T.wo_w = nxt(); T.wqkv_b = nxt(); T.wqkv_w = nxt();
    }
    P.t_ln_b = nxt(); P.t_ln_g = nxt();
    P.t_w1_b = nxt(); P.t_w1_w = nxt();
    P.t_w2_b = nxt(); P.t_w2_w = nxt();
  }
  const float *tor_l1_b = nxt(), *tor_l1_w = nxt();
  const float *tor_l2_b = nxt(), *tor_l2_w = nxt();
  const float *tor_lf_b = nxt(), *tor_lf_w = nxt();

  // ---- workspace ----
  char* wsb = (char*)d_ws;
  size_t off = 0;
  auto alloc = [&](size_t nf) -> float* {
    float* p = (float*)(wsb + off);
    off = (off + nf * sizeof(float) + 255) & ~(size_t)255;
    return p;
  };
  float* zA   = alloc((size_t)BNN * CZ);
  float* zB   = alloc((size_t)BNN * CZ);
  float* s    = alloc((size_t)BN * CS);
  float* sl   = alloc((size_t)BN * CS);
  float* t    = alloc((size_t)BN * CS);
  float* ff   = alloc((size_t)BN * CS);
  float* to   = alloc((size_t)BN * CS);
  float* qb   = alloc((size_t)BN * Hh * CH);
  float* kb   = alloc((size_t)BN * Hh * CH);
  float* vb   = alloc((size_t)BN * Hh * CH);
  float* qkv  = alloc((size_t)BN * 3 * CS);
  float* qp   = alloc((size_t)BN * Hh * PQ * 3);
  float* kp   = alloc((size_t)BN * Hh * PQ * 3);
  float* vp   = alloc((size_t)BN * Hh * PV * 3);
  float* qn   = alloc((size_t)BN * Hh);
  float* kn   = alloc((size_t)BN * Hh);
  float* attn = alloc((size_t)Bb_ * Hh * Nn_ * Nn_);
  float* pqk  = alloc((size_t)Bb_ * Hh * Nn_ * Nn_);
  float* bwb  = alloc((size_t)BNN * Hh);
  float* tlog = alloc((size_t)Bb_ * THh * Nn_ * Nn_);
  float* opt  = alloc((size_t)BN * Hh * PV * 3);
  float* cat  = alloc((size_t)BN * CAT);
  float* rot  = alloc((size_t)BN * 9);
  float* quat = alloc((size_t)BN * 4);
  float* tr   = alloc((size_t)BN * 3);
  float* lft  = alloc((size_t)BN * CZ);
  float* rgt  = alloc((size_t)BN * CZ);
  float* un   = alloc((size_t)BN * 2);

  float* outp = (float*)d_out;

  // ---- setup ----
  hipMemcpyAsync(zA, edge_embed, (size_t)BNN * CZ * sizeof(float),
                 hipMemcpyDeviceToDevice, stream);
  hipMemcpyAsync(s, rigids_embed, (size_t)BN * CS * sizeof(float),
                 hipMemcpyDeviceToDevice, stream);
  k_init_rigids<<<(BN + 255) / 256, 256, 0, stream>>>(rigids_t, quat, tr, BN);

  float* zcur = zA;
  float* znxt = zB;
  const float qk_alpha  = 0.03608439182435161f;  // 1/sqrt(3*CH)
  const float t_alpha   = 0.125f;                // 1/sqrt(64)

  for (int b = 0; b < NB; ++b) {
    BlockP& P = blk[b];
    // rot from quat
    k_quat_rot<<<(BN + 255) / 256, 256, 0, stream>>>(quat, rot, BN);

    // ---------------- IPA ----------------
    k_layernorm<<<(BN + 7) / 8, 256, 0, stream>>>(s, P.ln_g, P.ln_b, sl, CS, BN);
    gemm(stream, 0, sl, CS,0,0, P.wq, Hh*CH,0,0, qb, Hh*CH,0,0, 0,0,0,0,0,0, BN, Hh*CH, CS, 1,1, 1.f);
    gemm(stream, 0, sl, CS,0,0, P.wk, Hh*CH,0,0, kb, Hh*CH,0,0, 0,0,0,0,0,0, BN, Hh*CH, CS, 1,1, 1.f);
    gemm(stream, 0, sl, CS,0,0, P.wv, Hh*CH,0,0, vb, Hh*CH,0,0, 0,0,0,0,0,0, BN, Hh*CH, CS, 1,1, 1.f);
    gemm(stream, 0, sl, CS,0,0, P.wqp, Hh*PQ*3,0,0, qp, Hh*PQ*3,0,0, 0,0,0,0,0,0, BN, Hh*PQ*3, CS, 1,1, 1.f);
    gemm(stream, 0, sl, CS,0,0, P.wkp, Hh*PQ*3,0,0, kp, Hh*PQ*3,0,0, 0,0,0,0,0,0, BN, Hh*PQ*3, CS, 1,1, 1.f);
    gemm(stream, 0, sl, CS,0,0, P.wvp, Hh*PV*3,0,0, vp, Hh*PV*3,0,0, 0,0,0,0,0,0, BN, Hh*PV*3, CS, 1,1, 1.f);
    k_apply_rot<<<(BN*Hh*PQ + 255)/256, 256, 0, stream>>>(qp, rot, tr, Hh*PQ, BN*Hh*PQ);
    k_apply_rot<<<(BN*Hh*PQ + 255)/256, 256, 0, stream>>>(kp, rot, tr, Hh*PQ, BN*Hh*PQ);
    k_apply_rot<<<(BN*Hh*PV + 255)/256, 256, 0, stream>>>(vp, rot, tr, Hh*PV, BN*Hh*PV);
    k_pointnorm<<<(BN*Hh + 255)/256, 256, 0, stream>>>(qp, qn, PQ, BN*Hh);
    k_pointnorm<<<(BN*Hh + 255)/256, 256, 0, stream>>>(kp, kn, PQ, BN*Hh);

    // logits: attn = (1/sqrt(3CH)) * q @ k^T   (batched over b,h)
    gemm(stream, GF_NT, qb, Hh*CH, (long)Nn_*Hh*CH, CH,
                        kb, Hh*CH, (long)Nn_*Hh*CH, CH,
                        attn, Nn_, (long)Hh*Nn_*Nn_, (long)Nn_*Nn_,
                        0, 0,0,0,0, 0, Nn_, Nn_, CH, Bb_, Hh, qk_alpha);
    // point logits: pqk = qp @ kp^T over (p,c), K=24
    gemm(stream, GF_NT, qp, Hh*PQ*3, (long)Nn_*Hh*PQ*3, PQ*3,
                        kp, Hh*PQ*3, (long)Nn_*Hh*PQ*3, PQ*3,
                        pqk, Nn_, (long)Hh*Nn_*Nn_, (long)Nn_*Nn_,
                        0, 0,0,0,0, 0, Nn_, Nn_, PQ*3, Bb_, Hh, 1.f);
    // pair bias: bwb = z @ wb  (M = B*N*N rows)
    gemm(stream, 0, zcur, CZ,0,0, P.wb, Hh,0,0, bwb, Hh,0,0,
         0, 0,0,0,0, 0, BNN, Hh, CZ, 1,1, 1.f);
    k_ipa_softmax<<<(Bb_*Hh*Nn_) / 4, 128, 0, stream>>>(attn, pqk, bwb, qn, kn,
                                                        P.gamma, res_mask);
    // o = attn @ v -> cat[:, 0:2048]
    gemm(stream, 0, attn, Nn_, (long)Hh*Nn_*Nn_, (long)Nn_*Nn_,
                  vb, Hh*CH, (long)Nn_*Hh*CH, CH,
                  cat, CAT, (long)Nn_*CAT, CH,
         0, 0,0,0,0, 0, Nn_, CH, Nn_, Bb_, Hh, 1.f);
    // opt = attn @ vp
    gemm(stream, 0, attn, Nn_, (long)Hh*Nn_*Nn_, (long)Nn_*Nn_,
                  vp, Hh*PV*3, (long)Nn_*Hh*PV*3, PV*3,
                  opt, Hh*PV*3, (long)Nn_*Hh*PV*3, PV*3,
         0, 0,0,0,0, 0, Nn_, PV*3, Nn_, Bb_, Hh, 1.f);
    // opair = attn[b,:,i,:] @ z[b,i] -> cat[:, 2432:3456]  (batched over b,i)
    gemm(stream, 0, attn, Nn_*Nn_, (long)Hh*Nn_*Nn_, Nn_,
                  zcur, CZ, (long)Nn_*Nn_*CZ, (long)Nn_*CZ,
                  cat + Hh*CH + Hh*PV*4, CZ, (long)Nn_*CAT, CAT,
         0, 0,0,0,0, 0, Hh, CZ, Nn_, Bb_, Nn_, 1.f);
    k_loc<<<(BN*Hh*PV + 255)/256, 256, 0, stream>>>(opt, rot, tr, cat, BN*Hh*PV);
    // s = (s + cat @ wo) * mask
    gemm(stream, GF_RES | GF_ROWMASK, cat, CAT,0,0, P.wo, CS,0,0, s, CS,0,0,
         0, s, CS,0,0, res_mask, BN, CS, CAT, 1,1, 1.f);

    // ---------------- transformer x2 ----------------
    hipMemcpyAsync(t, s, (size_t)BN * CS * sizeof(float),
                   hipMemcpyDeviceToDevice, stream);
    for (int l = 0; l < 2; ++l) {
      auto& T = P.tf[l];
      k_layernorm<<<(BN + 7) / 8, 256, 0, stream>>>(t, T.ln1_g, T.ln1_b, sl, CS, BN);
      gemm(stream, GF_BIAS, sl, CS,0,0, T.wqkv_w, 3*CS,0,0, qkv, 3*CS,0,0,
           T.wqkv_b, 0,0,0,0, 0, BN, 3*CS, CS, 1,1, 1.f);
      // logits (b,h): q @ k^T / sqrt(dh)
      gemm(stream, GF_NT, qkv, 3*CS, (long)Nn_*3*CS, 64,
                          qkv + CS, 3*CS, (long)Nn_*3*CS, 64,
                          tlog, Nn_, (long)THh*Nn_*Nn_, (long)Nn_*Nn_,
           0, 0,0,0,0, 0, Nn_, Nn_, 64, Bb_, THh, t_alpha);
      k_softmax_mask<<<(Bb_*THh*Nn_) / 4, 128, 0, stream>>>(tlog, res_mask, THh);
      // to = attn @ v
      gemm(stream, 0, tlog, Nn_, (long)THh*Nn_*Nn_, (long)Nn_*Nn_,
                    qkv + 2*CS, 3*CS, (long)Nn_*3*CS, 64,
                    to, CS, (long)Nn_*CS, 64,
           0, 0,0,0,0, 0, Nn_, 64, Nn_, Bb_, THh, 1.f);
      // t = t + to @ wo + b
      gemm(stream, GF_BIAS | GF_RES, to, CS,0,0, T.wo_w, CS,0,0, t, CS,0,0,
           T.wo_b, t, CS,0,0, 0, BN, CS, CS, 1,1, 1.f);
      // FFN
      k_layernorm<<<(BN + 7) / 8, 256, 0, stream>>>(t, T.ln2_g, T.ln2_b, sl, CS, BN);
      gemm(stream, GF_BIAS | GF_RELU, sl, CS,0,0, T.w1_w, FF,0,0, ff, FF,0,0,
           T.w1_b, 0,0,0,0, 0, BN, FF, CS, 1,1, 1.f);
      gemm(stream, GF_BIAS | GF_RES, ff, FF,0,0, T.w2_w, CS,0,0, t, CS,0,0,
           T.w2_b, t, CS,0,0, 0, BN, CS, FF, 1,1, 1.f);
    }
    // s = (s + t @ post) * mask
    gemm(stream, GF_RES | GF_ROWMASK, t, CS,0,0, P.post_w, CS,0,0, s, CS,0,0,
         0, s, CS,0,0, res_mask, BN, CS, CS, 1,1, 1.f);

    // ---------------- transition MLP ----------------
    k_layernorm<<<(BN + 7) / 8, 256, 0, stream>>>(s, P.t_ln_g, P.t_ln_b, sl, CS, BN);
    gemm(stream, GF_BIAS | GF_RELU, sl, CS,0,0, P.t_w1_w, CS,0,0, ff, CS,0,0,
         P.t_w1_b, 0,0,0,0, 0, BN, CS, CS, 1,1, 1.f);
    gemm(stream, GF_BIAS | GF_RES | GF_ROWMASK, ff, CS,0,0, P.t_w2_w, CS,0,0,
         s, CS,0,0, P.t_w2_b, s, CS,0,0, res_mask, BN, CS, CS, 1,1, 1.f);

    // ---------------- backbone update ----------------
    k_bb_update<<<(BN + 7) / 8, 256, 0, stream>>>(s, res_mask, fixed_mask,
                                                  P.bb_w, P.bb_b, rot, quat, tr, BN);

    // ---------------- edge transition ----------------
    if (P.has_edge) {
      k_layernorm<<<(BN + 7) / 8, 256, 0, stream>>>(s, P.e_lns_g, P.e_lns_b, sl, CS, BN);
      gemm(stream, GF_BIAS, sl, CS,0,0, P.e_left_w, CZ,0,0, lft, CZ,0,0,
           P.e_left_b, 0,0,0,0, 0, BN, CZ, CS, 1,1, 1.f);
      gemm(stream, GF_BIAS, sl, CS,0,0, P.e_right_w, CZ,0,0, rgt, CZ,0,0,
           P.e_right_b, 0,0,0,0, 0, BN, CZ, CS, 1,1, 1.f);
      k_edge_fused<<<BNN / 32, 256, 0, stream>>>(zcur, znxt, lft, rgt,
                                                 P.e_lnz_g, P.e_lnz_b,
                                                 P.e_w1_w, P.e_w1_b,
                                                 P.e_w2_w, P.e_w2_b, res_mask);
      float* tmp = zcur; zcur = znxt; znxt = tmp;
    }
  }

  // ---------------- torsion head + outputs ----------------
  gemm(stream, GF_BIAS | GF_RELU, s, CS,0,0, tor_l1_w, CS,0,0, ff, CS,0,0,
       tor_l1_b, 0,0,0,0, 0, BN, CS, CS, 1,1, 1.f);
  gemm(stream, GF_BIAS | GF_RES, ff, CS,0,0, tor_l2_w, CS,0,0, t, CS,0,0,
       tor_l2_b, s, CS,0,0, 0, BN, CS, CS, 1,1, 1.f);
  gemm(stream, GF_BIAS, t, CS,0,0, tor_lf_w, 2,0,0, un, 2,0,0,
       tor_lf_b, 0,0,0,0, 0, BN, 2, CS, 1,1, 1.f);

  k_final<<<(BN + 255) / 256, 256, 0, stream>>>(un, quat, tr, outp, BN);
  hipMemcpyAsync(outp + (size_t)BN * (2 + 7), s, (size_t)BN * CS * sizeof(float),
                 hipMemcpyDeviceToDevice, stream);
}
